// MoGLo_Net_84086869721618
// MI455X (gfx1250) — matrix operand order repaired and checked by reading, not compiled, not run on hardware
//
#include <hip/hip_runtime.h>
#include <hip/hip_bf16.h>

typedef __attribute__((ext_vector_type(16))) _Float16 v16h;
typedef __attribute__((ext_vector_type(8)))  float    v8f;
typedef __attribute__((ext_vector_type(4)))  int      v4i_t;

// ---- problem dims ----
#define BATCH 16
#define CCH   128
#define SPA   64
#define PADS  80      // 64 + 2*8
#define NPOS  13      // windows per spatial dim
#define PPOS  169     // NPOS*NPOS
#define NR1   39      // 3 * NPOS query rows/cols in padded space (7+5i+u)
#define NDISP 15
#define NBAND 585     // NR1 * NDISP valid banded entries per block
#define LDA   136     // LDS row pitch in halfs (breaks 256B bank cycle)
#define LDB   136

#if __has_builtin(__builtin_amdgcn_global_load_async_to_lds_b128) && \
    __has_builtin(__builtin_amdgcn_s_wait_asynccnt)
#define HAS_ASYNC_LDS 1
#else
#define HAS_ASYNC_LDS 0
#endif

// f1p/f2p: [B][80][80][128] fp16 channel-last
// G:       [B][39 r1i][15 dx][39 c1i * 15 dy] float (585 contiguous per block)

__device__ __forceinline__ int qcol(int i) { return 7 + 5 * (i / 3) + (i % 3); }

__device__ __forceinline__ v16h load_frag(const _Float16* p0, const _Float16* p1) {
    union U { uint4 q[2]; v16h v; } u;
    u.q[0] = *(const uint4*)p0;
    u.q[1] = *(const uint4*)p1;
    return u.v;
}

#if HAS_ASYNC_LDS
typedef __attribute__((address_space(1))) v4i_t gv4i_t;
typedef __attribute__((address_space(3))) v4i_t sv4i_t;
__device__ __forceinline__ void async_cp16(_Float16* lds_dst, const _Float16* gsrc) {
    __builtin_amdgcn_global_load_async_to_lds_b128(
        (gv4i_t*)gsrc, (sv4i_t*)lds_dst, 0, 0);
}
#endif

// ---------------- kernel 0: pad + fp32->fp16 convert, channel-last ----------------
// thread = (b, row, cchunk, col): coalesced 128B input runs, 16B vector writes.
__global__ __launch_bounds__(256) void pad_convert_kernel(
    const float* __restrict__ f1, const float* __restrict__ f2,
    _Float16* __restrict__ f1p, _Float16* __restrict__ f2p) {
    const size_t half = (size_t)BATCH * PADS * PADS * (CCH / 8);
    size_t tid = (size_t)blockIdx.x * blockDim.x + threadIdx.x;
    if (tid >= 2 * half) return;
    int which = tid >= half;
    size_t o = which ? tid - half : tid;
    int col = (int)(o % PADS); o /= PADS;
    int cc  = (int)(o & 15);   o >>= 4;
    int row = (int)(o % PADS);
    int b   = (int)(o / PADS);
    int sr = row - 8, sc = col - 8;
    union { uint4 q; _Float16 h[8]; } u;
    if (sr >= 0 && sr < SPA && sc >= 0 && sc < SPA) {
        const float* src = which ? f2 : f1;
        const float* base = src + (((size_t)b * CCH + cc * 8) * SPA + sr) * SPA + sc;
#pragma unroll
        for (int k = 0; k < 8; ++k) u.h[k] = (_Float16)base[(size_t)k * SPA * SPA];
    } else {
        u.q = make_uint4(0u, 0u, 0u, 0u);
    }
    _Float16* dst = (which ? f2p : f1p) +
        (((size_t)(b * PADS + row) * PADS + col) * CCH + cc * 8);
    *(uint4*)dst = u.q;
}

// ---------------- kernel 1: row-pair correlation GEMM on WMMA ----------------
// block = (b, r1i, dx); D[48x80] = A[48x128] x B[128x80], K = channels.
// Only 9 of 15 tiles intersect the +-7 band; results staged in LDS, then a
// branch-free gather writes the 585 banded entries contiguously to G.
__global__ __launch_bounds__(128) void corr_wmma_kernel(
    const _Float16* __restrict__ f1p, const _Float16* __restrict__ f2p,
    float* __restrict__ G) {
    __shared__ _Float16 As[48 * LDA];
    __shared__ _Float16 Bs[80 * LDB];
    __shared__ float    Dt[9][16 * 17];   // 9 band tiles, row pitch 17

    int id = blockIdx.x;
    int dx  = id % NDISP; id /= NDISP;
    int r1i = id % NR1;
    int b   = id / NR1;
    int r1 = qcol(r1i);        // padded query row
    int r2 = r1 - 7 + dx;      // padded ROI row, always in [0,76]
    int t = threadIdx.x;

    const _Float16* f1row = f1p + ((size_t)(b * PADS + r1) * PADS) * CCH;
    const _Float16* f2row = f2p + ((size_t)(b * PADS + r2) * PADS) * CCH;

#if HAS_ASYNC_LDS
    for (int idx = t; idx < NR1 * 16; idx += 128) {           // A: 39 real rows
        int m = idx >> 4, j = idx & 15;
        async_cp16(&As[m * LDA + j * 8], f1row + (size_t)qcol(m) * CCH + j * 8);
    }
    for (int idx = t; idx < 80 * 16; idx += 128) {            // B: 80 ROI cols
        int n = idx >> 4, j = idx & 15;
        async_cp16(&Bs[n * LDB + j * 8], f2row + (size_t)n * CCH + j * 8);
    }
    for (int idx = t; idx < 9 * 16; idx += 128) {             // zero-pad rows 39..47
        int m = NR1 + (idx >> 4), j = idx & 15;
        *(uint4*)(&As[m * LDA + j * 8]) = make_uint4(0u, 0u, 0u, 0u);
    }
    __builtin_amdgcn_s_wait_asynccnt(0);
#else
    for (int idx = t; idx < 48 * 16; idx += 128) {
        int m = idx >> 4, j = idx & 15;
        uint4 val = make_uint4(0u, 0u, 0u, 0u);
        if (m < NR1) val = *(const uint4*)(f1row + (size_t)qcol(m) * CCH + j * 8);
        *(uint4*)(&As[m * LDA + j * 8]) = val;
    }
    for (int idx = t; idx < 80 * 16; idx += 128) {
        int n = idx >> 4, j = idx & 15;
        *(uint4*)(&Bs[n * LDB + j * 8]) = *(const uint4*)(f2row + (size_t)n * CCH + j * 8);
    }
#endif
    __syncthreads();

    int wave = t >> 5, lane = t & 31;
    int lrow = lane & 15;
    int hi   = lane >> 4;  // lane group selects contiguous K half (k = 16*hi + s), same for A and B

    // band tiles j=0..8 -> (mtile, ntile); tile id in Dt equals j.
    for (int j = wave; j < 9; j += 4) {
        int mtile = (j >= 3) + (j >= 7);
        int ntile = j - 2 * mtile;
        v8f acc = {};
        const _Float16* arow = &As[(mtile * 16 + lrow) * LDA + hi * 16];
        const _Float16* brow = &Bs[(ntile * 16 + lrow) * LDB + hi * 16];
#pragma unroll
        for (int kk = 0; kk < 4; ++kk) {               // K = 128 = 4 x 32
            v16h af = load_frag(arow + kk * 32, arow + kk * 32 + 8);
            v16h bf = load_frag(brow + kk * 32, brow + kk * 32 + 8);
            acc = __builtin_amdgcn_wmma_f32_16x16x32_f16(
                false, af, false, bf, (short)0, acc, false, false);
        }
        float* dtile = Dt[j];
#pragma unroll
        for (int i = 0; i < 8; ++i)                    // M = i + 8*hi, N = lrow
            dtile[(i + 8 * hi) * 17 + lrow] = acc[i];
    }
    __syncthreads();

    // branch-free banded gather -> contiguous coalesced store (585 floats)
    float* gbase = G + (size_t)((b * NR1 + r1i) * NDISP + dx) * NBAND;
    for (int e = t; e < NBAND; e += 128) {
        int mg = e / NDISP, dy = e % NDISP;
        int q2 = qcol(mg) - 7 + dy;                    // in [0,76]
        int tid2 = (mg >> 4) * 2 + (q2 >> 4);          // band tile id
        gbase[e] = Dt[tid2][(mg & 15) * 17 + (q2 & 15)];
    }
}

// ---------------- kernel 2: norms, 3x3 box combine, transpose, upsample ----------------
__global__ __launch_bounds__(256) void finalize_kernel(
    const _Float16* __restrict__ f1p, const _Float16* __restrict__ f2p,
    const float* __restrict__ G, float* __restrict__ out) {
    __shared__ float red[256];
    __shared__ float sq[17 * 18];
    __shared__ float m2a[225];
    __shared__ float corr[15 * 16];   // stored transposed: corr[dy][dx]
    __shared__ float m1s;

    int bp = blockIdx.x;
    int b = bp / PPOS, p = bp % PPOS;
    int pi = p / NPOS, pj = p % NPOS;
    int t = threadIdx.x;

    // ||w1||: L2 over 3x3x128 query patch
    float s = 0.0f;
    for (int idx = t; idx < 9 * CCH; idx += 256) {
        int c = idx & (CCH - 1);
        int uv = idx >> 7;
        int u = uv / 3, v = uv % 3;
        float x = (float)f1p[(((size_t)b * PADS + (7 + 5 * pi + u)) * PADS + (7 + 5 * pj + v)) * CCH + c];
        s += x * x;
    }
    red[t] = s; __syncthreads();
    for (int off = 128; off > 0; off >>= 1) {
        if (t < off) red[t] += red[t + off];
        __syncthreads();
    }
    if (t == 0) m1s = sqrtf(red[0]);
    __syncthreads();

    // sq[x][y] = sum_c roi2^2
    for (int e = t; e < 17 * 17; e += 256) {
        int x = e / 17, y = e % 17;
        const _Float16* src = f2p + (((size_t)b * PADS + (5 * pi + x)) * PADS + (5 * pj + y)) * CCH;
        float a = 0.0f;
        for (int c = 0; c < CCH; c += 8) {
            union { uint4 q; _Float16 h[8]; } u;
            u.q = *(const uint4*)(src + c);
#pragma unroll
            for (int k = 0; k < 8; ++k) { float f = (float)u.h[k]; a += f * f; }
        }
        sq[x * 18 + y] = a;
    }
    __syncthreads();

    // m2 = sqrt(3x3 box-sum)
    for (int e = t; e < 225; e += 256) {
        int dx = e / NDISP, dy = e % NDISP;
        float bsum = 0.0f;
#pragma unroll
        for (int u = 0; u < 3; ++u)
#pragma unroll
            for (int v = 0; v < 3; ++v) bsum += sq[(dx + u) * 18 + (dy + v)];
        m2a[e] = sqrtf(bsum);
    }
    __syncthreads();

    // dot = sum_{u,v} Gram; normalize; store transposed
    float inv_m1 = 1.0f / (m1s + 1e-6f);
    for (int e = t; e < 225; e += 256) {
        int dx = e / NDISP, dy = e % NDISP;
        float d = 0.0f;
#pragma unroll
        for (int u = 0; u < 3; ++u)
#pragma unroll
            for (int v = 0; v < 3; ++v) {
                size_t gi = (size_t)((b * NR1 + (3 * pi + u)) * NDISP + dx) * NBAND
                          + (3 * pj + v) * NDISP + dy;
                d += G[gi];
            }
        corr[dy * 16 + dx] = d * inv_m1 / (m2a[e] + 1e-6f);
    }
    __syncthreads();

    // bilinear upsample 15 -> 16, align_corners=True
    {
        int i = t >> 4, j = t & 15;
        const float sc = 14.0f / 15.0f;
        float si = i * sc, sj = j * sc;
        int i0 = (int)si, j0 = (int)sj;
        int i1 = i0 + 1 > 14 ? 14 : i0 + 1;
        int j1 = j0 + 1 > 14 ? 14 : j0 + 1;
        float ti = si - (float)i0, tj = sj - (float)j0;
        float v00 = corr[i0 * 16 + j0], v01 = corr[i0 * 16 + j1];
        float v10 = corr[i1 * 16 + j0], v11 = corr[i1 * 16 + j1];
        float val = v00 * (1.0f - ti) * (1.0f - tj) + v01 * (1.0f - ti) * tj
                  + v10 * ti * (1.0f - tj) + v11 * ti * tj;
        out[(((size_t)b * PPOS + p) * 16 + i) * 16 + j] = val;
    }
}

extern "C" void kernel_launch(void* const* d_in, const int* in_sizes, int n_in,
                              void* d_out, int out_size, void* d_ws, size_t ws_size,
                              hipStream_t stream) {
    (void)in_sizes; (void)n_in; (void)out_size; (void)ws_size;
    const float* f1 = (const float*)d_in[0];
    const float* f2 = (const float*)d_in[1];
    float* out = (float*)d_out;

    const size_t padded_elems = (size_t)BATCH * PADS * PADS * CCH;   // 13.1M halfs
    _Float16* f1p = (_Float16*)d_ws;
    _Float16* f2p = f1p + padded_elems;
    float* G = (float*)((char*)d_ws + 2 * padded_elems * sizeof(_Float16)); // 21.9 MB

    // 1) pad + convert to fp16 channel-last
    {
        size_t total = 2 * (size_t)BATCH * PADS * PADS * (CCH / 8);
        int blocks = (int)((total + 255) / 256);
        pad_convert_kernel<<<blocks, 256, 0, stream>>>(f1, f2, f1p, f2p);
    }
    // 2) WMMA row-pair correlation GEMM (band tiles only)
    {
        int blocks = BATCH * NR1 * NDISP;   // 9360
        corr_wmma_kernel<<<blocks, 128, 0, stream>>>(f1p, f2p, G);
    }
    // 3) normalize + transpose + upsample
    {
        finalize_kernel<<<BATCH * PPOS, 256, 0, stream>>>(f1p, f2p, G, out);
    }
}
